// TorchTest_76304388980778
// MI455X (gfx1250) — matrix operand order repaired and checked by reading, not compiled, not run on hardware
//
#include <hip/hip_runtime.h>
#include <hip/hip_bf16.h>
#include <math.h>

typedef __attribute__((ext_vector_type(16))) _Float16 v16h;
typedef __attribute__((ext_vector_type(8)))  float    v8f;
typedef __attribute__((ext_vector_type(4)))  unsigned su4;
typedef __attribute__((ext_vector_type(8)))  unsigned su8;

#define S4_B  8
#define S4_H  256
#define S4_N2 32

// Butterfly all-reduce across a wave32 (all lanes end with the sum).
__device__ __forceinline__ float wave_allreduce(float v) {
    v += __shfl_xor(v, 16, 32);
    v += __shfl_xor(v,  8, 32);
    v += __shfl_xor(v,  4, 32);
    v += __shfl_xor(v,  2, 32);
    v += __shfl_xor(v,  1, 32);
    return v;
}

// --------------------------------------------------------------------------
// Kernel 1: ZOH discretization. Per (h,n): w = exp(dt*A),
// Ct_s = 2 * C_s * (exp(dt*A)-1)/A  (conjugate-pair doubling folded in).
// pars layout: [h*N2+n][8] = {w_re, w_im, Ct0_re, Ct0_im, Ct1_re, Ct1_im, 0, 0}
// --------------------------------------------------------------------------
__global__ void s4_prep(const float* __restrict__ log_dt,
                        const float* __restrict__ A_re, const float* __restrict__ A_im,
                        const float* __restrict__ C_re, const float* __restrict__ C_im,
                        float* __restrict__ pars) {
    int t = blockIdx.x * blockDim.x + threadIdx.x;          // H*N2 threads
    if (t >= S4_H * S4_N2) return;
    int h = t / S4_N2;
    float dt = expf(log_dt[h]);
    float ar = A_re[t], ai = A_im[t];
    float dr = dt * ar, di = dt * ai;
    float em = expf(dr);
    float wr = em * cosf(di), wi = em * sinf(di);
    float nr = wr - 1.0f, ni = wi;                          // exp(dtA) - 1
    float den = ar * ar + ai * ai;
    float fr = (nr * ar + ni * ai) / den;                   // (exp(dtA)-1)/A
    float fi = (ni * ar - nr * ai) / den;
    float* P = pars + (size_t)t * 8;
    P[0] = wr; P[1] = wi;
#pragma unroll
    for (int s = 0; s < 2; ++s) {
        float cr = C_re[s * S4_H * S4_N2 + t];
        float ci = C_im[s * S4_H * S4_N2 + t];
        P[2 + s * 2] = 2.0f * (cr * fr - ci * fi);
        P[3 + s * 2] = 2.0f * (cr * fi + ci * fr);
    }
    P[6] = 0.0f; P[7] = 0.0f;
}

// --------------------------------------------------------------------------
// Kernel 2: W_out fp32 -> f16 (row-major [512][256]) for WMMA A-matrix.
// --------------------------------------------------------------------------
__global__ void s4_wcvt(const float* __restrict__ W, _Float16* __restrict__ wf, int n) {
    int t = blockIdx.x * blockDim.x + threadIdx.x;
    if (t < n) wf[t] = (_Float16)W[t];
}

// --------------------------------------------------------------------------
// Kernel 3: bidirectional diagonal-SSM scan. One wave per (b,h); lane = state n.
//   backward pass:  t=0; for p=L-1..0: y_b[p]=2Re(Ct1·t); t = w*t + u[p]
//   forward  pass:  s=0; for p=0..L-1: s = w*s + u[p]; y_f[p]=2Re(Ct0·s)
//   y = y_f + y_b + D*u ; exact-erf GELU ; store f16 in [B][L][H] layout.
// --------------------------------------------------------------------------
__global__ void __launch_bounds__(256) s4_scan(
        const float* __restrict__ x, const float* __restrict__ pars,
        const float* __restrict__ Dv, float* __restrict__ yb,
        _Float16* __restrict__ yg, int L) {
    int wid  = blockIdx.x * (blockDim.x >> 5) + (threadIdx.x >> 5);
    int lane = threadIdx.x & 31;
    int b = wid / S4_H, h = wid % S4_H;
    const float* P = pars + ((size_t)h * S4_N2 + lane) * 8;
    float wr = P[0], wi = P[1];
    float c0r = P[2], c0i = P[3], c1r = P[4], c1i = P[5];
    float Dh = Dv[h];
    const float* xb  = x + (size_t)b * S4_H + h;            // x[(p*B+b)*H + h]
    float*       ybp = yb + ((size_t)b * S4_H + h) * (size_t)L;
    const size_t xs = (size_t)S4_B * S4_H;

    // ---- backward (anti-causal) pass ----
    float sr = 0.0f, si = 0.0f, slot = 0.0f;
    for (int p = L - 1; p >= 0; --p) {
        float t = wave_allreduce(c1r * sr - c1i * si);
        if ((p & 31) == lane) slot = t;
        if ((p & 31) == 0) ybp[p + lane] = slot;            // coalesced 32-wide
        float u   = xb[(size_t)p * xs];
        float nsr = wr * sr - wi * si + u;
        si = wr * si + wi * sr;
        sr = nsr;
    }
    asm volatile("s_wait_storecnt 0" ::: "memory");         // own stores visible

    // ---- forward (causal) pass + skip + GELU ----
    sr = 0.0f; si = 0.0f;
    for (int p = 0; p < L; ++p) {
        float u   = xb[(size_t)p * xs];
        float nsr = wr * sr - wi * si + u;
        si = wr * si + wi * sr;
        sr = nsr;
        float t = wave_allreduce(c0r * sr - c0i * si);
        float y = t + ybp[p] + Dh * u;
        float g = 0.5f * y * (1.0f + erff(y * 0.70710678118654752f));
        if ((p & 31) == lane) slot = g;
        if ((p & 31) == 31)
            yg[((size_t)b * L + (p - 31 + lane)) * S4_H + h] = (_Float16)slot;
    }
}

// --------------------------------------------------------------------------
// Kernel 4: 1x1 conv (M=512, K=256, N=B*L) with v_wmma_f32_16x16x32_f16 + GLU.
// The 16x256 f16 activation tile (contiguous 8KB in [B][L][H] layout) is DMA'd
// into LDS by the Tensor Data Mover (tensor_load_to_lds, 1-D descriptor),
// completion tracked with TENSORcnt. 8 waves; wave w owns output-channel tiles
// pt = 2w, 2w+1 (rows pt*16+m of the "a" half pair with 256+pt*16+m for GLU).
// --------------------------------------------------------------------------
__global__ void __launch_bounds__(256) s4_glu_gemm(
        const _Float16* __restrict__ yg, const _Float16* __restrict__ wf,
        const float* __restrict__ bias, float* __restrict__ out, int L) {
    __shared__ uint4 sh[512];                               // 16 cols x 256 h f16 = 8KB
    int tilesPerB = L >> 4;
    int ct    = blockIdx.x;
    int b     = ct / tilesPerB;
    int lbase = (ct % tilesPerB) << 4;
    int tid   = threadIdx.x;

    // ---- TDM: stage activation tile Global -> LDS (one issue, wave 0) ----
    if (tid < 32) {
        unsigned long long ga =
            (unsigned long long)(const void*)(yg + ((size_t)b * L + lbase) * S4_H);
        unsigned ldsOff = (unsigned)(unsigned long long)(void*)sh; // LDS aperture: addr[31:0]
        su4 g0;
        g0.x = 1u;                                          // count=1, user mode
        g0.y = ldsOff;                                      // lds_addr (bytes)
        g0.z = (unsigned)(ga & 0xFFFFFFFFull);              // global_addr[31:0]
        g0.w = (unsigned)(ga >> 32) | (2u << 30);           // global_addr[56:32] | type=2
        su8 g1;
        g1.s0 = 1u << 16;                                   // data_size=1 (2 bytes/elem)
        g1.s1 = 4096u << 16;                                // tensor_dim0 = 4096 elems (lo16)
        g1.s2 = 1u << 16;                                   // tensor_dim0 hi=0 | tensor_dim1=1
        g1.s3 = 4096u << 16;                                // tile_dim0 = 4096 (1-D tile)
        g1.s4 = 0u;                                         // tile_dim1=0, tile_dim2=0
        g1.s5 = 4096u;                                      // tensor_dim0_stride lo32
        g1.s6 = 0u;                                         // stride0 hi | stride1 lo
        g1.s7 = 0u;
        asm volatile("tensor_load_to_lds %0, %1" :: "s"(g0), "s"(g1) : "memory");
        __builtin_amdgcn_s_wait_tensorcnt(0);               // data resident in LDS
    }
    __syncthreads();

    int wave = tid >> 5, lane = tid & 31;
    int n = lane & 15, hiL = lane >> 4;
    const unsigned* shU = (const unsigned*)sh;
    const unsigned* wU  = (const unsigned*)wf;

    v8f accA[2], accG[2];
    accA[0] = (v8f)(0.0f); accA[1] = (v8f)(0.0f);
    accG[0] = (v8f)(0.0f); accG[1] = (v8f)(0.0f);

    union U { v16h v; unsigned u[8]; };

#pragma unroll
    for (int kt = 0; kt < 8; ++kt) {                        // K = 256 in steps of 32
        int kb = kt * 32;
        // B fragment (32x16, K=h chunk, N=l column). Contiguous k order used here:
        // lanes 0-15 hold K=kb..kb+15, lanes 16-31 hold K=kb+16..kb+31, pairs/VGPR.
        U bf;
#pragma unroll
        for (int vv = 0; vv < 8; ++vv)
            bf.u[vv] = shU[n * 128 + (kb >> 1) + hiL * 8 + vv];
#pragma unroll
        for (int p = 0; p < 2; ++p) {
            int pt = wave * 2 + p;
            int oa = pt * 16 + n;                           // m = lane&15 (row of A)
            int og = 256 + oa;
            // A fragment (16x32 f16), same K order as B: lanes 0-15 hold K=0..15,
            // lanes 16-31 hold K=16..31, pairs/VGPR.
            U aA, aG;
#pragma unroll
            for (int vv = 0; vv < 8; ++vv) {
                int k = kb + hiL * 16 + 2 * vv;
                aA.u[vv] = wU[oa * 128 + (k >> 1)];
                aG.u[vv] = wU[og * 128 + (k >> 1)];
            }
            accA[p] = __builtin_amdgcn_wmma_f32_16x16x32_f16(
                false, aA.v, false, bf.v, (short)0, accA[p], false, false);
            accG[p] = __builtin_amdgcn_wmma_f32_16x16x32_f16(
                false, aG.v, false, bf.v, (short)0, accG[p], false, false);
        }
    }

    // Epilogue: C/D layout -> element (m,n): m = hiL*8 + reg, n = lane&15.  GLU.
#pragma unroll
    for (int p = 0; p < 2; ++p) {
        int pt = wave * 2 + p;
#pragma unroll
        for (int r = 0; r < 8; ++r) {
            int m  = hiL * 8 + r;
            int hh = pt * 16 + m;
            float va = accA[p][r] + bias[hh];
            float vg = accG[p][r] + bias[256 + hh];
            float o  = va / (1.0f + expf(-vg));             // a * sigmoid(g)
            out[((size_t)b * S4_H + hh) * L + lbase + n] = o;
        }
    }
}

// --------------------------------------------------------------------------
// Host launcher. Workspace layout (bytes):
//   [0, 256K)        pars  (H*N2*8 f32)
//   [256K, 512K)     W f16 (512*256)
//   [512K, +BHL*4)   y_b   f32  [B][H][L]
//   then             y_gelu f16 [B][L][H]
// --------------------------------------------------------------------------
extern "C" void kernel_launch(void* const* d_in, const int* in_sizes, int n_in,
                              void* d_out, int out_size, void* d_ws, size_t ws_size,
                              hipStream_t stream) {
    const float* x      = (const float*)d_in[0];
    const float* log_dt = (const float*)d_in[1];
    const float* A_re   = (const float*)d_in[2];
    const float* A_im   = (const float*)d_in[3];
    const float* C_re   = (const float*)d_in[4];
    const float* C_im   = (const float*)d_in[5];
    const float* Dv     = (const float*)d_in[6];
    const float* W      = (const float*)d_in[7];
    const float* bout   = (const float*)d_in[8];
    float* out = (float*)d_out;

    int L = in_sizes[0] / (S4_B * S4_H);

    char* ws = (char*)d_ws;
    float*    pars = (float*)ws;
    _Float16* wf   = (_Float16*)(ws + 262144);
    float*    yb   = (float*)(ws + 524288);
    _Float16* yg   = (_Float16*)(ws + 524288 + (size_t)S4_B * S4_H * (size_t)L * 4);

    s4_prep<<<(S4_H * S4_N2 + 255) / 256, 256, 0, stream>>>(log_dt, A_re, A_im,
                                                            C_re, C_im, pars);
    s4_wcvt<<<(2 * S4_H * S4_H + 255) / 256, 256, 0, stream>>>(W, wf, 2 * S4_H * S4_H);
    s4_scan<<<(S4_B * S4_H) / 8, 256, 0, stream>>>(x, pars, Dv, yb, yg, L);
    s4_glu_gemm<<<(S4_B * L) / 16, 256, 0, stream>>>(yg, wf, bout, out, L);
}
